// Qwen3Attention_40278203302130
// MI455X (gfx1250) — matrix operand order repaired and checked by reading, not compiled, not run on hardware
//
#include <hip/hip_runtime.h>
#include <hip/hip_bf16.h>

typedef __attribute__((ext_vector_type(16))) _Float16 v16h;
typedef __attribute__((ext_vector_type(8)))  _Float16 v8h;
typedef __attribute__((ext_vector_type(8)))  float    v8f;

#define WMMA_F16(a,b,c) __builtin_amdgcn_wmma_f32_16x16x32_f16(false,(a),false,(b),(short)0,(c),false,false)

// ---- optional CDNA5 async global->LDS path (compiles either way) ----
#if defined(__has_builtin)
#if __has_builtin(__builtin_amdgcn_global_load_async_to_lds_b128) && \
    __has_builtin(__builtin_amdgcn_s_wait_asynccnt)
#define USE_ASYNC_LDS 1
#endif
#endif

#ifdef USE_ASYNC_LDS
typedef int v4i_t __attribute__((ext_vector_type(4)));
typedef __attribute__((address_space(1))) v4i_t* g_v4i_p;
typedef __attribute__((address_space(3))) v4i_t* l_v4i_p;
#define ASYNC_LD_B128(g, l, off) \
    __builtin_amdgcn_global_load_async_to_lds_b128((g_v4i_p)(g), (l_v4i_p)(l), (off), 0)
#define ASYNC_WAIT() __builtin_amdgcn_s_wait_asynccnt(0)
#endif

constexpr int Bc = 2, Sc = 2048, Hc = 4096, NHc = 32, NKVc = 8, HDc = 128, Gc = 4;
constexpr float EPSc = 1e-6f;
constexpr float SCALEc = 0.08838834764831845f; // 128^-0.5

__device__ __forceinline__ v16h pack16(v8h lo, v8h hi) {
    v16h r;
#pragma unroll
    for (int i = 0; i < 8; ++i) { r[i] = lo[i]; r[i + 8] = hi[i]; }
    return r;
}

// ---------------------------------------------------------------- convert ---
__global__ void cvt_f32_f16(const float* __restrict__ in, _Float16* __restrict__ out, int n) {
    int i = blockIdx.x * blockDim.x + threadIdx.x;
    if (i < n) out[i] = (_Float16)in[i];
}

// ------------------------------------------------------------------- GEMM ---
// C[M,N] = A[M,K] * B[K,N], A/B f16 row-major, C f32. Block = 256 thr (8 waves),
// tile 64x128, BK=32. A staged in LDS (async if available); B-frags loaded
// directly from global (lane l holds B column n=l&15, k in table order).
__global__ void __launch_bounds__(256)
gemm_f16f32(const _Float16* __restrict__ A, const _Float16* __restrict__ Bm,
            float* __restrict__ C, int M, int N, int K) {
    __shared__ _Float16 As[64][40];          // padded stride vs bank conflicts
    const int t    = threadIdx.x;
    const int lane = t & 31;
    const int w    = t >> 5;
    const int h    = lane >> 4;
    const int nl   = lane & 15;
    const int m0   = blockIdx.y * 64;
    const int n0   = blockIdx.x * 128;
    const int wm   = (w >> 1) * 16;          // wave row tile (within block)
    const int wn   = n0 + (w & 1) * 64;      // wave col base (4 x 16 tiles)

    const int ar = t >> 2;                   // staging: row 0..63
    const int ac = (t & 3) * 8;              // staging: col start

    v8f acc[4] = {};

    for (int k0 = 0; k0 < K; k0 += 32) {
        __syncthreads();
        const _Float16* ap = &A[(size_t)(m0 + ar) * K + k0 + ac];
#ifdef USE_ASYNC_LDS
        ASYNC_LD_B128(ap, &As[ar][ac], 0);
        ASYNC_WAIT();
#else
        *(v8h*)&As[ar][ac] = *(const v8h*)ap;
#endif
        __syncthreads();

        // A fragment (16x32)
        v16h af = pack16(*(const v8h*)&As[wm + nl][h * 8],
                         *(const v8h*)&As[wm + nl][16 + h * 8]);

        // B fragments direct from global
        const int kb = k0 + 8 * h;
        const _Float16* bp = Bm + (size_t)kb * N + wn + nl;
        if (k0 + 32 < K) __builtin_prefetch(bp + (size_t)32 * N, 0, 1);
#pragma unroll
        for (int c = 0; c < 4; ++c) {
            v16h bf;
#pragma unroll
            for (int e = 0; e < 8; ++e) {
                bf[e]     = bp[(size_t)e * N + 16 * c];
                bf[e + 8] = bp[(size_t)(16 + e) * N + 16 * c];
            }
            acc[c] = WMMA_F16(af, bf, acc[c]);
        }
    }

    const int row0 = m0 + wm + 8 * h;        // C layout: VGPR i -> row i+8h
#pragma unroll
    for (int c = 0; c < 4; ++c)
#pragma unroll
        for (int i = 0; i < 8; ++i)
            C[(size_t)(row0 + i) * N + wn + 16 * c + nl] = acc[c][i];
}

// ---------------------------------------------------- RMSNorm + RoPE (f16) ---
// X: (B,S,nh,HD) f32 -> Y: (B,nh,S,HD) f16. One block of 128 = one (b,s,head).
__global__ void __launch_bounds__(128)
norm_rope(const float* __restrict__ X, const float* __restrict__ wn,
          const float* __restrict__ cosb, const float* __restrict__ sinb,
          _Float16* __restrict__ Y, int nh) {
    __shared__ float xs[128];
    __shared__ float red[4];
    const int d   = threadIdx.x;
    const int bid = blockIdx.x;
    const int hh  = bid % nh;
    const int s   = (bid / nh) % Sc;
    const int b   = bid / (nh * Sc);

    float x = X[((size_t)(b * Sc + s) * nh + hh) * HDc + d];
    float v = x * x;
#pragma unroll
    for (int off = 16; off; off >>= 1) v += __shfl_xor(v, off, 32);
    if ((d & 31) == 0) red[d >> 5] = v;
    __syncthreads();
    const float var = (red[0] + red[1] + red[2] + red[3]) * (1.0f / HDc);
    const float xn  = x * rsqrtf(var + EPSc) * wn[d];
    xs[d] = xn;
    __syncthreads();
    const float rot = (d < 64) ? -xs[d + 64] : xs[d - 64];
    const size_t ci = (size_t)(b * Sc + s) * HDc + d;
    const float o   = xn * cosb[ci] + rot * sinb[ci];
    Y[((size_t)(b * nh + hh) * Sc + s) * HDc + d] = (_Float16)o;
}

// --------------------------------------------- V transpose f32->(B,KV,S,D) ---
__global__ void v_trans(const float* __restrict__ V, _Float16* __restrict__ Y, int n) {
    int i = blockIdx.x * blockDim.x + threadIdx.x;
    if (i >= n) return;
    const int d  = i & 127;
    const int hh = (i >> 7) & 7;
    const int s  = (i >> 10) & 2047;
    const int b  = i >> 21;
    Y[((size_t)(b * NKVc + hh) * Sc + s) * HDc + d] = (_Float16)V[i];
}

// -------------------------------------------------------- flash attention ---
// Grid: (S/128, B*NH). Block = 8 waves; wave w owns the 16-query tile
// q in [q0+16w, q0+16w+16). K tile (32 keys) transposed into LDS, V tile
// staged row-major in LDS (async if available), P bounced LDS->A-frag.
__global__ void __launch_bounds__(256)
flash_attn(const _Float16* __restrict__ Q, const _Float16* __restrict__ Kg,
           const _Float16* __restrict__ Vg, _Float16* __restrict__ AO) {
    __shared__ _Float16 Kt[128][32];     // K tile transposed: [d][j]
    __shared__ _Float16 Vt[32][136];     // V tile row-major, padded stride
    __shared__ _Float16 Pl[8][16][32];   // per-wave P staging

    const int t    = threadIdx.x;
    const int lane = t & 31;
    const int w    = t >> 5;
    const int h    = lane >> 4;
    const int nl   = lane & 15;
    const int bh   = blockIdx.y;
    const int b    = bh / NHc;
    const int hh   = bh % NHc;
    const int kv   = hh / Gc;
    const int q0   = blockIdx.x * 128;
    const int qt   = q0 + w * 16;

    const _Float16* Qb = Q  + (size_t)(b * NHc  + hh) * Sc * HDc;
    const _Float16* Kb = Kg + (size_t)(b * NKVc + kv) * Sc * HDc;
    const _Float16* Vb = Vg + (size_t)(b * NKVc + kv) * Sc * HDc;

    // Q fragments (16x128 = 4 A-frags), kept in registers
    v16h qf[4];
#pragma unroll
    for (int dc = 0; dc < 4; ++dc) {
        const _Float16* qp = Qb + (size_t)(qt + nl) * HDc + dc * 32 + h * 8;
        qf[dc] = pack16(*(const v8h*)qp, *(const v8h*)(qp + 16));
    }

    float rmax[8], rsum[8];
    v8f Of[8];
    v8f zero = {};
#pragma unroll
    for (int i = 0; i < 8; ++i) { rmax[i] = -3.0e38f; rsum[i] = 0.f; Of[i] = zero; }

    const int jEnd = q0 + 128;
    const int sr = t >> 3;            // staging: key row 0..31
    const int sd = (t & 7) * 16;      // staging: d start

    for (int j0 = 0; j0 < jEnd; j0 += 32) {
        __syncthreads();
        {   // stage K tile transposed + V tile row-major
            const _Float16* kp = Kb + (size_t)(j0 + sr) * HDc + sd;
            const _Float16* vp = Vb + (size_t)(j0 + sr) * HDc + sd;
#ifdef USE_ASYNC_LDS
            ASYNC_LD_B128(vp, &Vt[sr][sd], 0);
            ASYNC_LD_B128(vp, &Vt[sr][sd], 16);
#else
            *(v8h*)&Vt[sr][sd]     = *(const v8h*)vp;
            *(v8h*)&Vt[sr][sd + 8] = *(const v8h*)(vp + 8);
#endif
            v8h k0v = *(const v8h*)kp;
            v8h k1v = *(const v8h*)(kp + 8);
#pragma unroll
            for (int e = 0; e < 8; ++e) {
                Kt[sd + e][sr]     = k0v[e];
                Kt[sd + 8 + e][sr] = k1v[e];
            }
#ifdef USE_ASYNC_LDS
            ASYNC_WAIT();
#endif
        }
        __syncthreads();

        if (j0 <= qt + 15) {
            // ---- scores: S = Q * K^T (two 16x16 frags, K-dim = 128) ----
            v8f s0 = {}; v8f s1 = {};
#pragma unroll
            for (int dc = 0; dc < 4; ++dc) {
                const int drow = dc * 32 + 8 * h;     // B k base within chunk (table order)
                v16h b0, b1;
#pragma unroll
                for (int e = 0; e < 8; ++e) {
                    b0[e]     = Kt[drow + e][nl];
                    b0[e + 8] = Kt[drow + 16 + e][nl];
                    b1[e]     = Kt[drow + e][16 + nl];
                    b1[e + 8] = Kt[drow + 16 + e][16 + nl];
                }
                s0 = WMMA_F16(qf[dc], b0, s0);
                s1 = WMMA_F16(qf[dc], b1, s1);
            }

            // ---- online softmax (rows live in 16-lane halves) ----
            float p0[8], p1[8], alpha[8];
#pragma unroll
            for (int i = 0; i < 8; ++i) {
                const int q = qt + i + 8 * h;
                float a  = s0[i] * SCALEc;
                float bb = s1[i] * SCALEc;
                if (j0 + nl      > q) a  = -3.0e38f;
                if (j0 + 16 + nl > q) bb = -3.0e38f;
                float mx = fmaxf(a, bb);
#pragma unroll
                for (int off = 8; off; off >>= 1) mx = fmaxf(mx, __shfl_xor(mx, off, 32));
                const float nm = fmaxf(rmax[i], mx);
                const float al = __expf(rmax[i] - nm);
                const float e0 = __expf(a  - nm);
                const float e1 = __expf(bb - nm);
                float ss = e0 + e1;
#pragma unroll
                for (int off = 8; off; off >>= 1) ss += __shfl_xor(ss, off, 32);
                rsum[i]  = rsum[i] * al + ss;
                rmax[i]  = nm;
                alpha[i] = al;
                p0[i] = e0; p1[i] = e1;
            }
#pragma unroll
            for (int fi = 0; fi < 8; ++fi)
#pragma unroll
                for (int i = 0; i < 8; ++i) Of[fi][i] *= alpha[i];

            // ---- P: C-layout -> LDS -> A-frag ----
#pragma unroll
            for (int i = 0; i < 8; ++i) {
                Pl[w][i + 8 * h][nl]      = (_Float16)p0[i];
                Pl[w][i + 8 * h][nl + 16] = (_Float16)p1[i];
            }
            asm volatile("s_wait_dscnt 0" ::: "memory");
            const _Float16* pp = &Pl[w][nl][8 * h];
            v16h pf = pack16(*(const v8h*)pp, *(const v8h*)(pp + 16));

            // ---- O += P * V (V B-frags from LDS tile) ----
            const int kloc = 8 * h;
            const _Float16* vrow = &Vt[kloc][0];
#pragma unroll
            for (int fd = 0; fd < 8; ++fd) {
                v16h vf;
#pragma unroll
                for (int e = 0; e < 8; ++e) {
                    vf[e]     = vrow[e * 136 + fd * 16 + nl];
                    vf[e + 8] = vrow[(16 + e) * 136 + fd * 16 + nl];
                }
                Of[fd] = WMMA_F16(pf, vf, Of[fd]);
            }
        }
    }

    // epilogue: O /= l, write (b, q, hh*128 + d) f16
    float inv[8];
#pragma unroll
    for (int i = 0; i < 8; ++i) inv[i] = 1.0f / rsum[i];
#pragma unroll
    for (int fd = 0; fd < 8; ++fd)
#pragma unroll
        for (int i = 0; i < 8; ++i) {
            const int q = qt + i + 8 * h;
            AO[(size_t)(b * Sc + q) * (NHc * HDc) + hh * HDc + fd * 16 + nl] =
                (_Float16)(Of[fd][i] * inv[i]);
        }
}

// ------------------------------------------------------------------ launch ---
extern "C" void kernel_launch(void* const* d_in, const int* in_sizes, int n_in,
                              void* d_out, int out_size, void* d_ws, size_t ws_size,
                              hipStream_t stream) {
    const float* hs   = (const float*)d_in[0];
    const float* cosb = (const float*)d_in[1];
    const float* sinb = (const float*)d_in[2];
    const float* Wq   = (const float*)d_in[3];
    const float* Wk   = (const float*)d_in[4];
    const float* Wv   = (const float*)d_in[5];
    const float* Wo   = (const float*)d_in[6];
    const float* qnw  = (const float*)d_in[7];
    const float* knw  = (const float*)d_in[8];

    const size_t BS = (size_t)Bc * Sc;        // 4096 tokens
    char* p = (char*)d_ws;
    auto alloc = [&](size_t bytes) { void* r = (void*)p; p += (bytes + 255) & ~(size_t)255; return r; };

    _Float16* Xh  = (_Float16*)alloc(BS * Hc * 2);
    _Float16* Wqh = (_Float16*)alloc((size_t)Hc * 4096 * 2);
    _Float16* Wkh = (_Float16*)alloc((size_t)Hc * 1024 * 2);
    _Float16* Wvh = (_Float16*)alloc((size_t)Hc * 1024 * 2);
    _Float16* Woh = (_Float16*)alloc((size_t)4096 * Hc * 2);
    float*    Qf  = (float*)   alloc(BS * 4096 * 4);
    float*    Kf  = (float*)   alloc(BS * 1024 * 4);
    float*    Vf  = (float*)   alloc(BS * 1024 * 4);
    _Float16* Qh  = (_Float16*)alloc(BS * 4096 * 2);
    _Float16* Kh  = (_Float16*)alloc(BS * 1024 * 2);
    _Float16* Vh  = (_Float16*)alloc(BS * 1024 * 2);
    _Float16* AOh = (_Float16*)alloc(BS * 4096 * 2);

    // f32 -> f16 conversions
    {
        int n = (int)(BS * Hc);
        cvt_f32_f16<<<n / 256, 256, 0, stream>>>(hs, Xh, n);
        n = Hc * 4096;
        cvt_f32_f16<<<n / 256, 256, 0, stream>>>(Wq, Wqh, n);
        cvt_f32_f16<<<n / 256, 256, 0, stream>>>(Wo, Woh, n);
        n = Hc * 1024;
        cvt_f32_f16<<<n / 256, 256, 0, stream>>>(Wk, Wkh, n);
        cvt_f32_f16<<<n / 256, 256, 0, stream>>>(Wv, Wvh, n);
    }

    // QKV projections (WMMA GEMMs), block tile 64x128
    gemm_f16f32<<<dim3(4096 / 128, 64), 256, 0, stream>>>(Xh, Wqh, Qf, 4096, 4096, 4096);
    gemm_f16f32<<<dim3(1024 / 128, 64), 256, 0, stream>>>(Xh, Wkh, Kf, 4096, 1024, 4096);
    gemm_f16f32<<<dim3(1024 / 128, 64), 256, 0, stream>>>(Xh, Wvh, Vf, 4096, 1024, 4096);

    // RMSNorm + RoPE -> (B,h,S,D) f16 ; V transpose
    norm_rope<<<Bc * Sc * NHc, 128, 0, stream>>>(Qf, qnw, cosb, sinb, Qh, NHc);
    norm_rope<<<Bc * Sc * NKVc, 128, 0, stream>>>(Kf, knw, cosb, sinb, Kh, NKVc);
    {
        int n = (int)(BS * 1024);
        v_trans<<<n / 256, 256, 0, stream>>>(Vf, Vh, n);
    }

    // causal GQA flash attention
    flash_attn<<<dim3(Sc / 128, Bc * NHc), 256, 0, stream>>>(Qh, Kh, Vh, AOh);

    // output projection -> fp32 d_out
    gemm_f16f32<<<dim3(4096 / 128, 64), 256, 0, stream>>>(AOh, Woh, (float*)d_out, 4096, 4096, 4096);

    (void)in_sizes; (void)n_in; (void)out_size; (void)ws_size;
}
